// MoELayer_11003706212976
// MI455X (gfx1250) — matrix operand order repaired and checked by reading, not compiled, not run on hardware
//
#include <hip/hip_runtime.h>
#include <stddef.h>

// ---------------- problem constants ----------------
#define TOK     4096            // B*S tokens
#define HDIM    1024
#define H2      2048
#define NEXP    8
#define PADROWS 8448            // >= 8192 routed slots + 8*31 tile padding (32-row tiles)
#define NTILES  264             // max 32-row tiles across experts

typedef __bf16 bf16;
typedef __attribute__((ext_vector_type(4)))  float  v4f;
typedef __attribute__((ext_vector_type(4)))  __bf16 v4bf;
typedef __attribute__((ext_vector_type(8)))  __bf16 v8bf;
typedef __attribute__((ext_vector_type(16))) __bf16 v16bf;
typedef __attribute__((ext_vector_type(8)))  float  v8f;

__device__ __forceinline__ v8f wmma_bf16(v16bf a, v16bf b, v8f c) {
  // D = A(16x32 bf16) * B(32x16 bf16) + C(16x16 f32)
  return __builtin_amdgcn_wmma_f32_16x16x32_bf16(
      false, a, false, b, (short)0, c, false, false);
}

// A-fragment from row-major LDS row (pre-offset by g*8): runs [kk+g*8..+7],[kk+16+g*8..+7]
__device__ __forceinline__ v16bf load_a(const bf16* p) {
  v8bf a0 = *(const v8bf*)(p);
  v8bf a1 = *(const v8bf*)(p + 16);
  return __builtin_shufflevector(a0, a1, 0, 1, 2, 3, 4, 5, 6, 7,
                                 8, 9, 10, 11, 12, 13, 14, 15);
}

// gfx1250 async direct-to-LDS copy (16B), tracked by ASYNCcnt
__device__ __forceinline__ void async_ld16(unsigned lds_byte_off, const void* g) {
  asm volatile("global_load_async_to_lds_b128 %0, %1, off"
               :: "v"(lds_byte_off), "v"((unsigned long long)(size_t)g)
               : "memory");
}
__device__ __forceinline__ void wait_async0() {
  asm volatile("s_wait_asynccnt 0x0" ::: "memory");
}

// ---------------- init ----------------
__global__ __launch_bounds__(256) void init_meta_kernel(int* __restrict__ counts,
                                                        int* __restrict__ slot_token,
                                                        float* __restrict__ slot_gate) {
  int i = blockIdx.x * 256 + threadIdx.x;
  if (i < PADROWS) { slot_token[i] = 0; slot_gate[i] = 0.f; }
  if (i < NEXP) counts[i] = 0;
}

// ---------------- fp32 -> bf16 convert of activations ----------------
__global__ __launch_bounds__(256) void convert_x_kernel(const float* __restrict__ x,
                                                        bf16* __restrict__ xb) {
  int i = blockIdx.x * 256 + threadIdx.x;       // one float4 per thread
  v4f v = ((const v4f*)x)[i];
  v4bf o;
  o.x = (bf16)v.x; o.y = (bf16)v.y; o.z = (bf16)v.z; o.w = (bf16)v.w;
  ((v4bf*)xb)[i] = o;
}

// ---------------- pack weights into WMMA B-fragment layout ----------------
// layout: [e][nt][kt][lane][i], i=0..15 contiguous bf16 per lane (32B)
//   K = kt*32 + (lane>>4)*8 + (i<8 ? i : i+8) ,  N = nt*16 + (lane&15)
__global__ __launch_bounds__(256) void pack_w1_kernel(const float* __restrict__ W1,
                                                      bf16* __restrict__ W1p) {
  unsigned tid = blockIdx.x * 256 + threadIdx.x;          // 8*128*32*32 = 1,048,576
  unsigned lane = tid & 31, kt = (tid >> 5) & 31, nt = (tid >> 10) & 127, e = tid >> 17;
  unsigned g = lane >> 4, n = nt * 16 + (lane & 15);
  const float* src = W1 + ((size_t)e * HDIM + kt * 32 + g * 8) * H2 + n;
  v8bf lo, hi;
#pragma unroll
  for (int i = 0; i < 8; ++i) lo[i] = (bf16)src[(size_t)i * H2];
#pragma unroll
  for (int i = 0; i < 8; ++i) hi[i] = (bf16)src[(size_t)(i + 16) * H2];
  v8bf* dst = (v8bf*)(W1p + (size_t)tid * 16);
  dst[0] = lo; dst[1] = hi;
}

__global__ __launch_bounds__(256) void pack_w2_kernel(const float* __restrict__ W2,
                                                      bf16* __restrict__ W2p) {
  unsigned tid = blockIdx.x * 256 + threadIdx.x;          // 8*64*64*32 = 1,048,576
  unsigned lane = tid & 31, kt = (tid >> 5) & 63, nt = (tid >> 11) & 63, e = tid >> 17;
  unsigned g = lane >> 4, n = nt * 16 + (lane & 15);
  const float* src = W2 + ((size_t)e * H2 + kt * 32 + g * 8) * HDIM + n;
  v8bf lo, hi;
#pragma unroll
  for (int i = 0; i < 8; ++i) lo[i] = (bf16)src[(size_t)i * HDIM];
#pragma unroll
  for (int i = 0; i < 8; ++i) hi[i] = (bf16)src[(size_t)(i + 16) * HDIM];
  v8bf* dst = (v8bf*)(W2p + (size_t)tid * 16);
  dst[0] = lo; dst[1] = hi;
}

// ---------------- router: logits, top-2, softmax ----------------
__global__ __launch_bounds__(256) void router_kernel(const float* __restrict__ x,
                                                     const float* __restrict__ Wr,
                                                     const float* __restrict__ br,
                                                     int* __restrict__ counts,
                                                     int* __restrict__ tk_idx,
                                                     float* __restrict__ tk_w) {
  __shared__ float WrS[HDIM * NEXP];                      // 32 KB
  for (int i = threadIdx.x; i < HDIM * NEXP; i += 256) WrS[i] = Wr[i];
  __syncthreads();
  int t = blockIdx.x * 256 + threadIdx.x;
  const float* xr = x + (size_t)t * HDIM;
  float acc[NEXP] = {};
  for (int h = 0; h < HDIM; ++h) {
    float xv = xr[h];
#pragma unroll
    for (int e = 0; e < NEXP; ++e) acc[e] = fmaf(xv, WrS[h * NEXP + e], acc[e]);
  }
  float best = -INFINITY, sec = -INFINITY; int bi = 0, si = 0;
#pragma unroll
  for (int e = 0; e < NEXP; ++e) {
    float l = acc[e] + br[e];
    if (l > best)      { sec = best; si = bi; best = l; bi = e; }
    else if (l > sec)  { sec = l; si = e; }
  }
  float e1  = __expf(sec - best);
  float inv = 1.0f / (1.0f + e1);
  tk_idx[t * 2 + 0] = bi; tk_w[t * 2 + 0] = inv;
  tk_idx[t * 2 + 1] = si; tk_w[t * 2 + 1] = e1 * inv;
  atomicAdd(&counts[bi], 1);
  atomicAdd(&counts[si], 1);
}

// ---------------- per-expert 32-padded offsets ----------------
__global__ void offsets_kernel(const int* __restrict__ counts, int* __restrict__ offs,
                               int* __restrict__ cursor) {
  int base = 0;
  for (int e = 0; e < NEXP; ++e) {
    offs[e] = base; cursor[e] = base;
    base += ((counts[e] + 31) >> 5) << 5;
  }
  offs[NEXP] = base;                                      // total padded rows
}

// ---------------- scatter tokens into expert slots ----------------
__global__ __launch_bounds__(256) void scatter_kernel(const int* __restrict__ tk_idx,
                                                      const float* __restrict__ tk_w,
                                                      int* __restrict__ cursor,
                                                      int* __restrict__ slot_token,
                                                      float* __restrict__ slot_gate,
                                                      int* __restrict__ tok2slot) {
  int i = blockIdx.x * 256 + threadIdx.x;                 // T*2 assignments
  int e = tk_idx[i];
  int slot = atomicAdd(&cursor[e], 1);
  slot_token[slot] = i >> 1;
  slot_gate[slot]  = tk_w[i];
  tok2slot[i]      = slot;
}

// ---------------- GEMM1: h = relu(X W1 + b1), 32 rows/WG, bf16 WMMA ----------------
#define XS_STRIDE 1032                                    // 1024 + 8 pad (bank-friendly)
__global__ __launch_bounds__(256) void gemm1_kernel(const bf16* __restrict__ Xbf,
                                                    const bf16* __restrict__ W1p,
                                                    const float* __restrict__ b1,
                                                    const int* __restrict__ offs,
                                                    const int* __restrict__ slot_token,
                                                    bf16* __restrict__ Hbf) {
  extern __shared__ __align__(16) char smem[];
  bf16* Xs = (bf16*)smem;                                 // 32 x XS_STRIDE bf16
  int rt = blockIdx.x, total = offs[NEXP], row0 = rt * 32;
  if (row0 >= total) return;
  int e = 0;
#pragma unroll
  for (int i = 1; i < NEXP; ++i) if (row0 >= offs[i]) e = i;

  unsigned lb = __builtin_amdgcn_groupstaticsize();       // dynamic-LDS base offset
  for (int c = threadIdx.x; c < 32 * 128; c += 256) {     // gather 32 rows x 1024 bf16
    int r = c >> 7, col = (c & 127) << 3;
    int tok = slot_token[row0 + r];
    async_ld16(lb + (unsigned)(r * XS_STRIDE + col) * 2,
               Xbf + (size_t)tok * HDIM + col);
  }
  wait_async0();
  __syncthreads();

  int lane = threadIdx.x & 31, wave = threadIdx.x >> 5;
  int m = lane & 15, g = lane >> 4;
  const bf16* arow0 = Xs + m * XS_STRIDE + g * 8;         // M-tile 0
  const bf16* arow1 = Xs + (m + 16) * XS_STRIDE + g * 8;  // M-tile 1

#pragma unroll 1
  for (int grp = 0; grp < 4; ++grp) {                     // wave owns 16 N-tiles, 4 at a time
    int ntb = wave * 16 + grp * 4;
    v8f acc0[4] = {}, acc1[4] = {};
    const bf16* bp[4];
#pragma unroll
    for (int u = 0; u < 4; ++u)
      bp[u] = W1p + (size_t)(e * 128 + ntb + u) * 16384 + lane * 16;
    for (int kt = 0; kt < 32; ++kt) {
      v16bf a0 = load_a(arow0 + kt * 32);
      v16bf a1 = load_a(arow1 + kt * 32);
#pragma unroll
      for (int u = 0; u < 4; ++u) {
        v16bf b = *(const v16bf*)(bp[u] + kt * 512);      // one contiguous 32B fragment
        acc0[u] = wmma_bf16(a0, b, acc0[u]);
        acc1[u] = wmma_bf16(a1, b, acc1[u]);              // B reused across 2 M-tiles
      }
    }
#pragma unroll
    for (int u = 0; u < 4; ++u) {
      int col = (ntb + u) * 16 + m;
      float bias = b1[e * H2 + col];
#pragma unroll
      for (int r = 0; r < 8; ++r) {                       // C layout: M = r + 8*g
        int ra = row0 + r + 8 * g;
        int rb = row0 + 16 + r + 8 * g;
        Hbf[(size_t)ra * H2 + col] = (bf16)fmaxf(acc0[u][r] + bias, 0.0f);
        Hbf[(size_t)rb * H2 + col] = (bf16)fmaxf(acc1[u][r] + bias, 0.0f);
      }
    }
  }
}

// ---------------- GEMM2: yperm = gate * (h W2 + b2), 32 rows/WG ----------------
#define HS_STRIDE 2056                                    // 2048 + 8 pad
__global__ __launch_bounds__(256) void gemm2_kernel(const bf16* __restrict__ Hbf,
                                                    const bf16* __restrict__ W2p,
                                                    const float* __restrict__ b2,
                                                    const int* __restrict__ offs,
                                                    const float* __restrict__ slot_gate,
                                                    float* __restrict__ yperm) {
  extern __shared__ __align__(16) char smem[];
  bf16* Hs = (bf16*)smem;                                 // 32 x HS_STRIDE bf16
  int rt = blockIdx.x, total = offs[NEXP], row0 = rt * 32;
  if (row0 >= total) return;
  int e = 0;
#pragma unroll
  for (int i = 1; i < NEXP; ++i) if (row0 >= offs[i]) e = i;

  unsigned lb = __builtin_amdgcn_groupstaticsize();
  for (int c = threadIdx.x; c < 32 * 256; c += 256) {     // 32 rows x 2048 bf16
    int r = c >> 8, col = (c & 255) << 3;
    async_ld16(lb + (unsigned)(r * HS_STRIDE + col) * 2,
               Hbf + (size_t)(row0 + r) * H2 + col);
  }
  wait_async0();
  __syncthreads();

  int lane = threadIdx.x & 31, wave = threadIdx.x >> 5;
  int m = lane & 15, g = lane >> 4;
  const bf16* arow0 = Hs + m * HS_STRIDE + g * 8;
  const bf16* arow1 = Hs + (m + 16) * HS_STRIDE + g * 8;

  float gate0[8], gate1[8];
#pragma unroll
  for (int r = 0; r < 8; ++r) {
    gate0[r] = slot_gate[row0 + r + 8 * g];
    gate1[r] = slot_gate[row0 + 16 + r + 8 * g];
  }

#pragma unroll 1
  for (int grp = 0; grp < 2; ++grp) {                     // wave owns 8 N-tiles
    int ntb = wave * 8 + grp * 4;
    v8f acc0[4] = {}, acc1[4] = {};
    const bf16* bp[4];
#pragma unroll
    for (int u = 0; u < 4; ++u)
      bp[u] = W2p + (size_t)(e * 64 + ntb + u) * 32768 + lane * 16;
    for (int kt = 0; kt < 64; ++kt) {
      v16bf a0 = load_a(arow0 + kt * 32);
      v16bf a1 = load_a(arow1 + kt * 32);
#pragma unroll
      for (int u = 0; u < 4; ++u) {
        v16bf b = *(const v16bf*)(bp[u] + kt * 512);
        acc0[u] = wmma_bf16(a0, b, acc0[u]);
        acc1[u] = wmma_bf16(a1, b, acc1[u]);
      }
    }
#pragma unroll
    for (int u = 0; u < 4; ++u) {
      int col = (ntb + u) * 16 + m;
      float bias = b2[e * HDIM + col];
#pragma unroll
      for (int r = 0; r < 8; ++r) {
        int ra = row0 + r + 8 * g;
        int rb = row0 + 16 + r + 8 * g;
        yperm[(size_t)ra * HDIM + col] = gate0[r] * (acc0[u][r] + bias);
        yperm[(size_t)rb * HDIM + col] = gate1[r] * (acc1[u][r] + bias);
      }
    }
  }
}

// ---------------- final gather: out[t] = y[slot(t,0)] + y[slot(t,1)] ----------------
__global__ __launch_bounds__(256) void gather_kernel(const float* __restrict__ yperm,
                                                     const int* __restrict__ tok2slot,
                                                     float* __restrict__ out) {
  int i = blockIdx.x * 256 + threadIdx.x;                 // TOK*HDIM/4 float4 chunks
  int t = i >> 8, c = i & 255;                            // 256 float4 per token
  int s0 = tok2slot[t * 2], s1 = tok2slot[t * 2 + 1];
  v4f a = ((const v4f*)yperm)[(size_t)s0 * 256 + c];
  v4f b = ((const v4f*)yperm)[(size_t)s1 * 256 + c];
  ((v4f*)out)[i] = a + b;                                 // fixed order -> deterministic
}

// ---------------- host launcher ----------------
extern "C" void kernel_launch(void* const* d_in, const int* in_sizes, int n_in,
                              void* d_out, int out_size, void* d_ws, size_t ws_size,
                              hipStream_t stream) {
  (void)in_sizes; (void)n_in; (void)out_size; (void)ws_size;
  const float* x  = (const float*)d_in[0];
  const float* Wr = (const float*)d_in[1];
  const float* br = (const float*)d_in[2];
  const float* W1 = (const float*)d_in[3];
  const float* b1 = (const float*)d_in[4];
  const float* W2 = (const float*)d_in[5];
  const float* b2 = (const float*)d_in[6];
  float* out = (float*)d_out;

  char* ws = (char*)d_ws;
  bf16*  Xbf  = (bf16*)(ws);                              //  8,388,608 B
  bf16*  W1p  = (bf16*)(ws + 8388608);                    // 33,554,432 B
  bf16*  W2p  = (bf16*)(ws + 41943040);                   // 33,554,432 B
  bf16*  Hbf  = (bf16*)(ws + 75497472);                   // 34,603,008 B (8448 x 2048 bf16)
  float* Yp   = (float*)(ws + 110100480);                 // 34,603,008 B (8448 x 1024 f32)
  char*  meta = ws + 144703488;
  int*   counts     = (int*)(meta);
  int*   cursor     = (int*)(meta + 32);
  int*   offs       = (int*)(meta + 64);                  // 9 ints
  int*   tk_idx     = (int*)(meta + 128);
  float* tk_w       = (float*)(meta + 32896);
  int*   slot_token = (int*)(meta + 65664);
  float* slot_gate  = (float*)(meta + 99456);
  int*   tok2slot   = (int*)(meta + 133248);

  // allow >64KB dynamic LDS (WGP has 320KB)
  hipFuncSetAttribute((const void*)gemm1_kernel,
                      hipFuncAttributeMaxDynamicSharedMemorySize, 32 * XS_STRIDE * 2);
  hipFuncSetAttribute((const void*)gemm2_kernel,
                      hipFuncAttributeMaxDynamicSharedMemorySize, 32 * HS_STRIDE * 2);

  init_meta_kernel<<<33, 256, 0, stream>>>(counts, slot_token, slot_gate);
  convert_x_kernel<<<4096, 256, 0, stream>>>(x, Xbf);
  pack_w1_kernel<<<4096, 256, 0, stream>>>(W1, W1p);
  pack_w2_kernel<<<4096, 256, 0, stream>>>(W2, W2p);
  router_kernel<<<TOK / 256, 256, 0, stream>>>(x, Wr, br, counts, tk_idx, tk_w);
  offsets_kernel<<<1, 1, 0, stream>>>(counts, offs, cursor);
  scatter_kernel<<<TOK * 2 / 256, 256, 0, stream>>>(tk_idx, tk_w, cursor,
                                                    slot_token, slot_gate, tok2slot);
  gemm1_kernel<<<NTILES, 256, 32 * XS_STRIDE * 2, stream>>>(Xbf, W1p, b1, offs,
                                                            slot_token, Hbf);
  gemm2_kernel<<<NTILES, 256, 32 * HS_STRIDE * 2, stream>>>(Hbf, W2p, b2, offs,
                                                            slot_gate, Yp);
  gather_kernel<<<4096, 256, 0, stream>>>(Yp, tok2slot, out);
}
